// ICIN_21680994910812
// MI455X (gfx1250) — matrix operand order repaired and checked by reading, not compiled, not run on hardware
//
#include <hip/hip_runtime.h>
#include <hip/hip_bf16.h>

// ---------------------------------------------------------------------------
// Types
// ---------------------------------------------------------------------------
typedef __attribute__((ext_vector_type(16))) __bf16 bf16x16;
typedef __attribute__((ext_vector_type(8)))  float  v8f;
typedef unsigned int u32x4 __attribute__((ext_vector_type(4)));
typedef int          i32x8 __attribute__((ext_vector_type(8)));
typedef int          i32x4 __attribute__((ext_vector_type(4)));

union FragB { uint4 u[2]; bf16x16 v; };

// Tensor Data Mover availability (this toolchain: 6-arg builtin, clang-23).
#if defined(__has_builtin)
#if __has_builtin(__builtin_amdgcn_tensor_load_to_lds) && \
    __has_builtin(__builtin_amdgcn_s_wait_tensorcnt)
#define USE_TDM 1
#endif
#endif
#ifndef USE_TDM
#define USE_TDM 0
#endif

__device__ __forceinline__ float gelu_f(float x) {
    return 0.5f * x * (1.0f + erff(x * 0.7071067811865476f));
}
__device__ __forceinline__ float sigm(float x) {
    return 1.0f / (1.0f + expf(-x));
}
__device__ __forceinline__ __bf16 f2b(float x) {
    union { float f; unsigned u; } a; a.f = x;
    unsigned r = a.u + 0x7FFFu + ((a.u >> 16) & 1u);   // round-to-nearest-even
    union { unsigned short s; __bf16 b; } o; o.s = (unsigned short)(r >> 16);
    return o.b;
}

#if USE_TDM
// Issue one TDM 2-D tile load: rows x 32 bf16 elements, row stride = K elems,
// packed contiguously into LDS at lds_addr.  Per-wave instruction (EXEC
// ignored); tracked with TENSORcnt.  D# layout per ISA 08_async_tensor 8.3/8.4.
__device__ __forceinline__ void tdm_load_tile(const __bf16* gsrc, unsigned lds_addr,
                                              int rows, int rowlen_elems) {
    unsigned long long ga = (unsigned long long)(size_t)gsrc;
    u32x4 g0;
    g0.x = 1u;                                              // count=1 (valid), user mode
    g0.y = lds_addr;                                        // LDS byte address
    g0.z = (unsigned)(ga & 0xFFFFFFFFu);                    // global_addr[31:0]
    g0.w = (unsigned)((ga >> 32) & 0x01FFFFFFu) | (2u << 30); // addr[56:32] | type=2
    unsigned k = (unsigned)rowlen_elems;
    unsigned r = (unsigned)rows;
    i32x8 g1;
    g1[0] = (int)(1u << 16);                 // data_size=1 (2B); no multicast/pad/iter
    g1[1] = (int)(k << 16);                  // tensor_dim0[15:0]  @ bits[63:48]
    g1[2] = (int)((k >> 16) | (r << 16));    // tensor_dim0[31:16] | tensor_dim1[15:0]
    g1[3] = (int)((r >> 16) | (32u << 16));  // tensor_dim1[31:16] | tile_dim0=32
    g1[4] = (int)r;                          // tile_dim1=rows; tile_dim2=0
    g1[5] = (int)k;                          // tensor_dim0_stride[31:0]
    g1[6] = 0;                               // stride0 hi | tensor_dim1_stride lo
    g1[7] = 0;
    i32x4 z4 = {0, 0, 0, 0};
    i32x8 z8 = {0, 0, 0, 0, 0, 0, 0, 0};
    __builtin_amdgcn_tensor_load_to_lds(g0, g1, z4, z4, z8, 0);
}
#endif

// ---------------------------------------------------------------------------
// Generic tiled bf16 WMMA GEMM:  out[M,N] = act(A[M,K] @ W[N,K]^T + bias)
// BM=128, BN=64, BK=32.  8 waves; each wave owns a 32x32 macro-tile
// (2x2 v_wmma_f32_16x16x32_bf16 fragments).  Double-buffered LDS; next
// K-step tiles streamed in by the Tensor Data Mover (wave0: A, wave1: B)
// while all waves run WMMA on the current buffer.
// act: 0=none 1=GELU 2=BN+GELU 3=ReLU 4=sigmoid(sigmoid(r))*aux*scalev
// ---------------------------------------------------------------------------
#define BM 128
#define BN 64
#define BK 32
#define BNSCALE 0.9999950000374997f   /* 1/sqrt(1+1e-5) */

__global__ __launch_bounds__(256) void gemm_bf16_wmma(
    const __bf16* __restrict__ A, const __bf16* __restrict__ W,
    int M, int N, int K,
    const float* __restrict__ bias, const float* __restrict__ gamma,
    const float* __restrict__ beta, const float* __restrict__ aux,
    const float* __restrict__ scalev, int act,
    __bf16* __restrict__ outb, float* __restrict__ outf)
{
    __shared__ __align__(16) __bf16 sA[2][BM][BK];
    __shared__ __align__(16) __bf16 sB[2][BN][BK];

    const int tid  = threadIdx.x;
    const int lane = tid & 31;
    const int wave = tid >> 5;
    const int bm   = blockIdx.x * BM;
    const int bn   = blockIdx.y * BN;
    const int wm   = (wave >> 1) * 32;
    const int wn   = (wave & 1) * 32;
    const int nlo  = lane & 15;
    const int hi   = lane >> 4;

    v8f acc[2][2] = {};
    const int nk = K / BK;

    // Register-staged cooperative copy (fallback / non-TDM prologue):
    // 3 independent global_load_b128 then 3 ds_store_b128, fully unrolled.
    auto copy_tiles = [&](int buf, int kt) {
        const int k0 = kt * BK;
        const int row = tid >> 2, col = (tid & 3) * 8;      // row: 0..63
        uint4 a0 = *(const uint4*)&A[(size_t)(bm + row) * K + k0 + col];
        uint4 a1 = *(const uint4*)&A[(size_t)(bm + row + 64) * K + k0 + col];
        uint4 b0 = *(const uint4*)&W[(size_t)(bn + row) * K + k0 + col];
        *(uint4*)&sA[buf][row][col]      = a0;
        *(uint4*)&sA[buf][row + 64][col] = a1;
        *(uint4*)&sB[buf][row][col]      = b0;
    };

#if USE_TDM
    auto tdm_tiles = [&](int buf, int kt) {
        const int k0 = kt * BK;
        if (wave == 0)
            tdm_load_tile(&A[(size_t)bm * K + k0],
                          (unsigned)(size_t)&sA[buf][0][0], BM, K);
        else if (wave == 1)
            tdm_load_tile(&W[(size_t)bn * K + k0],
                          (unsigned)(size_t)&sB[buf][0][0], BN, K);
    };
    tdm_tiles(0, 0);
    if (wave < 2) __builtin_amdgcn_s_wait_tensorcnt(0);
#else
    copy_tiles(0, 0);
#endif
    __syncthreads();

    for (int kt = 0; kt < nk; ++kt) {
        const int buf = kt & 1;
#if USE_TDM
        if (kt + 1 < nk) tdm_tiles(buf ^ 1, kt + 1);   // async, overlaps WMMA below
#else
        if (kt + 1 < nk) copy_tiles(buf ^ 1, kt + 1);
#endif
        if (kt + 2 < nk) {  // global_prefetch_b8 into L2 for the tile after next
            __builtin_prefetch(&A[(size_t)(bm + (tid & 127)) * K + (kt + 2) * BK], 0, 1);
        }

        // Fragment loads per ISA 7.12.2 layouts.
        FragB af[2], bfr[2];
        const int kbA = hi * 8;    // A: lanes 0-15 -> K base 0, lanes 16-31 -> 8
        const int kbB = hi * 8;    // B: lanes 0-15 -> K base 0, lanes 16-31 -> 8
#pragma unroll
        for (int i = 0; i < 2; ++i) {
            const __bf16* p = &sA[buf][wm + 16 * i + nlo][kbA];
            af[i].u[0] = *(const uint4*)p;
            af[i].u[1] = *(const uint4*)(p + 16);
        }
#pragma unroll
        for (int j = 0; j < 2; ++j) {
            const __bf16* p = &sB[buf][wn + 16 * j + nlo][kbB];
            bfr[j].u[0] = *(const uint4*)p;
            bfr[j].u[1] = *(const uint4*)(p + 16);
        }
#pragma unroll
        for (int i = 0; i < 2; ++i)
#pragma unroll
            for (int j = 0; j < 2; ++j)
                acc[i][j] = __builtin_amdgcn_wmma_f32_16x16x32_bf16(
                    false, af[i].v, false, bfr[j].v, (short)0, acc[i][j], false, false);

#if USE_TDM
        if (kt + 1 < nk && wave < 2) __builtin_amdgcn_s_wait_tensorcnt(0);
#endif
        __syncthreads();
    }

    // Epilogue (C layout: VGPR r -> row r + 8*(lane>=16), col = lane&15)
#pragma unroll
    for (int i = 0; i < 2; ++i)
#pragma unroll
        for (int j = 0; j < 2; ++j)
#pragma unroll
            for (int r = 0; r < 8; ++r) {
                int m = bm + wm + 16 * i + r + (hi << 3);
                int n = bn + wn + 16 * j + nlo;
                float v = acc[i][j][r];
                if (bias) v += bias[n];
                if (act == 1)      v = gelu_f(v);
                else if (act == 2) v = gelu_f(v * BNSCALE * gamma[n] + beta[n]);
                else if (act == 3) v = fmaxf(v, 0.0f);
                else if (act == 4) v = sigm(sigm(v)) * aux[(size_t)m * N + n] * scalev[n];
                if (outf) outf[(size_t)m * N + n] = v;
                if (outb) outb[(size_t)m * N + n] = f2b(v);
            }
}

// ---------------------------------------------------------------------------
// Flash-style attention over 4096x4096, D=768.
//   O = softmax(Q @ K^T * scale) @ V
// One block per 16 query rows; 128-key chunks; both GEMMs via WMMA.
// Q/K frags streamed from global (L2-resident, ~6MB); V staged transposed
// into LDS in 32-key sub-steps so B-frags are contiguous ds_load_b128.
// ---------------------------------------------------------------------------
__global__ __launch_bounds__(256) void attn_wmma(
    const __bf16* __restrict__ Q, const __bf16* __restrict__ Kp,
    const __bf16* __restrict__ Vp, float* __restrict__ Of,
    __bf16* __restrict__ Ob, int Btot, float scale)
{
    const int D = 768;
    __shared__ __align__(16) float  sS[16][128];
    __shared__ __align__(16) __bf16 sP[16][128];
    __shared__ __align__(16) __bf16 sVt[768][32];
    __shared__ float sMax[16], sSum[16], sAlpha[16];

    const int tid  = threadIdx.x;
    const int lane = tid & 31;
    const int wave = tid >> 5;
    const int m0   = blockIdx.x * 16;
    const int nlo  = lane & 15;
    const int hi   = lane >> 4;

    if (tid < 16) { sMax[tid] = -1e30f; sSum[tid] = 0.0f; }
    v8f acc[6] = {};
    __syncthreads();

    const int nchunks = Btot / 128;
    for (int j = 0; j < nchunks; ++j) {
        // ---- Phase A: S tile; wave w owns keys [j*128 + 16w, +16) ----
        v8f s = {};
        const int key = j * 128 + wave * 16 + nlo;
        for (int kk = 0; kk < D; kk += 32) {
            FragB a, b;
            const __bf16* pq = &Q[(size_t)(m0 + nlo) * D + kk + hi * 8];
            a.u[0] = *(const uint4*)pq;
            a.u[1] = *(const uint4*)(pq + 16);
            const __bf16* pk = &Kp[(size_t)key * D + kk + hi * 8];
            b.u[0] = *(const uint4*)pk;
            b.u[1] = *(const uint4*)(pk + 16);
            s = __builtin_amdgcn_wmma_f32_16x16x32_bf16(
                false, a.v, false, b.v, (short)0, s, false, false);
        }
#pragma unroll
        for (int r = 0; r < 8; ++r)
            sS[r + hi * 8][wave * 16 + nlo] = s[r] * scale;
        __syncthreads();

        // ---- online softmax: row r = tid>>4, 16 lanes per row ----
        {
            const int r = tid >> 4, slot = tid & 15;
            float oldM = sMax[r], oldS = sSum[r];
            float vals[8], cm = -1e30f;
#pragma unroll
            for (int c = 0; c < 8; ++c) {
                vals[c] = sS[r][slot * 8 + c];
                cm = fmaxf(cm, vals[c]);
            }
#pragma unroll
            for (int o = 1; o < 16; o <<= 1) cm = fmaxf(cm, __shfl_xor(cm, o, 32));
            float nm = fmaxf(oldM, cm);
            float alpha = expf(oldM - nm);
            float ps = 0.0f;
#pragma unroll
            for (int c = 0; c < 8; ++c) {
                float p = expf(vals[c] - nm);
                ps += p;
                sP[r][slot * 8 + c] = f2b(p);
            }
#pragma unroll
            for (int o = 1; o < 16; o <<= 1) ps += __shfl_xor(ps, o, 32);
            if (slot == 0) { sMax[r] = nm; sSum[r] = oldS * alpha + ps; sAlpha[r] = alpha; }
        }
        __syncthreads();

        // ---- rescale accumulators ----
        float a8[8];
#pragma unroll
        for (int r = 0; r < 8; ++r) a8[r] = sAlpha[r + hi * 8];
#pragma unroll
        for (int f = 0; f < 6; ++f)
#pragma unroll
            for (int r = 0; r < 8; ++r) acc[f][r] *= a8[r];

        // ---- Phase B: O += P @ V, 4 sub-steps of 32 keys ----
        for (int ks = 0; ks < 4; ++ks) {
            __syncthreads();  // previous sub-step's sVt readers done
            {
                const int base = j * 128 + ks * 32;
                for (int c = tid; c < 32 * 96; c += 256) {
                    int ky = c / 96, fc = c % 96;
                    uint4 d = *(const uint4*)&Vp[(size_t)(base + ky) * D + fc * 8];
                    const __bf16* e = (const __bf16*)&d;
#pragma unroll
                    for (int q = 0; q < 8; ++q) sVt[fc * 8 + q][ky] = e[q];
                }
            }
            __syncthreads();
            FragB a;
            const __bf16* pp = &sP[nlo][ks * 32 + hi * 8];
            a.u[0] = *(const uint4*)pp;
            a.u[1] = *(const uint4*)(pp + 16);
#pragma unroll
            for (int f = 0; f < 6; ++f) {
                const int feat = wave * 96 + f * 16 + nlo;
                FragB b;
                const __bf16* pv = &sVt[feat][hi * 8];
                b.u[0] = *(const uint4*)pv;
                b.u[1] = *(const uint4*)(pv + 16);
                acc[f] = __builtin_amdgcn_wmma_f32_16x16x32_bf16(
                    false, a.v, false, b.v, (short)0, acc[f], false, false);
            }
        }
        __syncthreads();  // sP consumed before next chunk rewrites it
    }

    float inv[8];
#pragma unroll
    for (int r = 0; r < 8; ++r) inv[r] = 1.0f / sSum[r + hi * 8];
#pragma unroll
    for (int f = 0; f < 6; ++f) {
        const int feat = wave * 96 + f * 16 + nlo;
#pragma unroll
        for (int r = 0; r < 8; ++r) {
            int m = m0 + r + hi * 8;
            float v = acc[f][r] * inv[r];
            Of[(size_t)m * D + feat] = v;
            Ob[(size_t)m * D + feat] = f2b(v);
        }
    }
}

// ---------------------------------------------------------------------------
// Small helper kernels
// ---------------------------------------------------------------------------
__global__ void k_cvt(const float* __restrict__ s, __bf16* __restrict__ d, int n) {
    for (int i = blockIdx.x * blockDim.x + threadIdx.x; i < n;
         i += gridDim.x * blockDim.x)
        d[i] = f2b(s[i]);
}
__global__ void k_gather_cvt(const float* __restrict__ s, __bf16* __restrict__ d,
                             int n, int stride, int off) {
    for (int i = blockIdx.x * blockDim.x + threadIdx.x; i < n;
         i += gridDim.x * blockDim.x)
        d[i] = f2b(s[(size_t)i * stride + off]);
}
__global__ void k_gather_f32(const float* __restrict__ s, float* __restrict__ d,
                             int n, int stride, int off) {
    for (int i = blockIdx.x * blockDim.x + threadIdx.x; i < n;
         i += gridDim.x * blockDim.x)
        d[i] = s[(size_t)i * stride + off];
}

// ca_h[B,24] = GELU(x1 @ ca_w1^T)  (K=384, too small for WMMA tiling)
__global__ void k_ca(const float* __restrict__ x1, const float* __restrict__ w,
                     float* __restrict__ out, int Btot) {
    int idx = blockIdx.x * blockDim.x + threadIdx.x;
    if (idx >= Btot * 24) return;
    int r = idx / 24, o = idx % 24;
    float d = 0.0f;
    for (int j = 0; j < 384; ++j)
        d += x1[(size_t)r * 384 + j] * w[o * 384 + j];
    out[idx] = gelu_f(d);
}

// Per-row FDA fusion: channel-attn (sigmoid(ca_h@w2^T)), spatial-attn (dot
// with center tap of 7x7), decision gate, sigma residual.  Block=128, 3
// channels per thread (IC=384).
__global__ __launch_bounds__(128) void k_fda_fuse(
    const float* __restrict__ x1, const float* __restrict__ cah,
    const float* __restrict__ caw2, const float* __restrict__ saw,
    const float* __restrict__ sab, const float* __restrict__ decw,
    const float* __restrict__ decb, const float* __restrict__ sigma,
    __bf16* __restrict__ x4)
{
    __shared__ float red[128];
    __shared__ float bc;
    const int r = blockIdx.x, tid = threadIdx.x, IC = 384;
    float x2[3], psa = 0.0f;
#pragma unroll
    for (int i = 0; i < 3; ++i) {
        int c = tid + i * 128;
        float d = 0.0f;
#pragma unroll
        for (int jj = 0; jj < 24; ++jj) d += cah[r * 24 + jj] * caw2[c * 24 + jj];
        float v = x1[(size_t)r * IC + c] * sigm(d);
        x2[i] = v;
        psa += v * saw[c];
    }
    red[tid] = psa; __syncthreads();
    for (int o = 64; o > 0; o >>= 1) { if (tid < o) red[tid] += red[tid + o]; __syncthreads(); }
    if (tid == 0) bc = sigm(red[0] + sab[0]);
    __syncthreads();
    const float s = bc;
    float pdc = 0.0f;
#pragma unroll
    for (int i = 0; i < 3; ++i) { x2[i] *= s; pdc += x2[i] * decw[tid + i * 128]; }
    __syncthreads();
    red[tid] = pdc; __syncthreads();
    for (int o = 64; o > 0; o >>= 1) { if (tid < o) red[tid] += red[tid + o]; __syncthreads(); }
    if (tid == 0) bc = gelu_f(red[0] + decb[0]);
    __syncthreads();
    const float xg = bc;
#pragma unroll
    for (int i = 0; i < 3; ++i) {
        int c = tid + i * 128;
        float v = x2[i];
        x4[(size_t)r * IC + c] = f2b(v + sigma[c] * (v - xg));
    }
}

// logits = (ct+ci) @ cls_w^T + cls_b; softmax over 3 classes
__global__ void k_head(const float* __restrict__ ct, const float* __restrict__ ci,
                       const float* __restrict__ w, const float* __restrict__ b,
                       float* __restrict__ out, int Btot) {
    for (int r = blockIdx.x * blockDim.x + threadIdx.x; r < Btot;
         r += gridDim.x * blockDim.x) {
        float lg[3];
        for (int l = 0; l < 3; ++l) {
            float d = b[l];
            for (int k = 0; k < 768; ++k)
                d += (ct[(size_t)r * 768 + k] + ci[(size_t)r * 768 + k]) * w[l * 768 + k];
            lg[l] = d;
        }
        float mx = fmaxf(lg[0], fmaxf(lg[1], lg[2]));
        float e0 = expf(lg[0] - mx), e1 = expf(lg[1] - mx), e2 = expf(lg[2] - mx);
        float is = 1.0f / (e0 + e1 + e2);
        out[r * 3 + 0] = e0 * is; out[r * 3 + 1] = e1 * is; out[r * 3 + 2] = e2 * is;
    }
}

// ---------------------------------------------------------------------------
// Host orchestration
// ---------------------------------------------------------------------------
extern "C" void kernel_launch(void* const* d_in, const int* in_sizes, int n_in,
                              void* d_out, int out_size, void* d_ws, size_t ws_size,
                              hipStream_t stream)
{
    (void)in_sizes; (void)n_in; (void)out_size; (void)ws_size;
    const int B = 4096, D = 768, IC = 384, S = 512;

    const float* text  = (const float*)d_in[0];
    const float* image = (const float*)d_in[1];
    const float* tl_w = (const float*)d_in[2];  const float* tl_b = (const float*)d_in[3];
    const float* il_w = (const float*)d_in[4];  const float* il_b = (const float*)d_in[5];
    const float* sda_wv = (const float*)d_in[10]; const float* sda_bv = (const float*)d_in[11];
    const float* sda_wo = (const float*)d_in[12]; const float* sda_bo = (const float*)d_in[13];
    const float* fda_w1 = (const float*)d_in[14]; const float* fda_b1 = (const float*)d_in[15];
    const float* bn1_g = (const float*)d_in[16];  const float* bn1_b = (const float*)d_in[17];
    const float* ca_w1 = (const float*)d_in[18];  const float* ca_w2 = (const float*)d_in[19];
    const float* sa_w = (const float*)d_in[20];   const float* sa_b = (const float*)d_in[21];
    const float* dec_w = (const float*)d_in[22];  const float* dec_b = (const float*)d_in[23];
    const float* sigma = (const float*)d_in[24];
    const float* fda_wf = (const float*)d_in[25]; const float* fda_bf = (const float*)d_in[26];
    const float* bn2_g = (const float*)d_in[27];  const float* bn2_b = (const float*)d_in[28];
    const float* dmi_wq = (const float*)d_in[29]; const float* dmi_bq = (const float*)d_in[30];
    const float* dmi_wk = (const float*)d_in[31]; const float* dmi_bk = (const float*)d_in[32];
    const float* dmi_wv = (const float*)d_in[33]; const float* dmi_bv = (const float*)d_in[34];
    const float* tg_w1 = (const float*)d_in[35];  const float* tg_b1 = (const float*)d_in[36];
    const float* tg_w2 = (const float*)d_in[37];  const float* tg_b2 = (const float*)d_in[38];
    const float* ig_w1 = (const float*)d_in[39];  const float* ig_b1 = (const float*)d_in[40];
    const float* ig_w2 = (const float*)d_in[41];  const float* ig_b2 = (const float*)d_in[42];
    const float* tscale = (const float*)d_in[43]; const float* iscale = (const float*)d_in[44];
    const float* cls_w = (const float*)d_in[45];  const float* cls_b = (const float*)d_in[46];

    char* ws = (char*)d_ws;
    size_t off = 0;
    auto allocB = [&](size_t elems) -> __bf16* {
        __bf16* p = (__bf16*)(ws + off);
        off += ((elems * 2 + 255) / 256) * 256;
        return p;
    };
    auto allocF = [&](size_t elems) -> float* {
        float* p = (float*)(ws + off);
        off += ((elems * 4 + 255) / 256) * 256;
        return p;
    };

    // bf16 weights
    __bf16* W_TL = allocB((size_t)D * D);
    __bf16* W_IL = allocB((size_t)D * D);
    __bf16* W_SV = allocB((size_t)S * D);
    __bf16* W_SO = allocB((size_t)D * S);
    __bf16* W_C1 = allocB((size_t)IC * D);
    __bf16* W_WF = allocB((size_t)D * IC);
    __bf16* W_Q  = allocB((size_t)D * D);
    __bf16* W_K  = allocB((size_t)D * D);
    __bf16* W_V  = allocB((size_t)D * D);
    __bf16* W_TG1 = allocB((size_t)D * D);
    __bf16* W_TG2 = allocB((size_t)D * D);
    __bf16* W_IG1 = allocB((size_t)D * D);
    __bf16* W_IG2 = allocB((size_t)D * D);
    float*  SAW = allocF(IC);

    // activations (with explicit reuse)
    __bf16* TB  = allocB((size_t)B * D);   // -> qp_t, then a_t_bf16
    __bf16* IB  = allocB((size_t)B * D);   // -> kp_im, then g1t
    __bf16* T1  = allocB((size_t)B * D);   // -> vp_im
    __bf16* IM1 = allocB((size_t)B * D);   // -> qp_im, then a_i_bf16
    __bf16* TV  = allocB((size_t)B * S);
    __bf16* T2  = allocB((size_t)B * D);
    __bf16* X4  = allocB((size_t)B * IC);
    __bf16* IM2 = allocB((size_t)B * D);
    __bf16* KPT = allocB((size_t)B * D);   // -> g1i
    __bf16* VPT = allocB((size_t)B * D);
    float*  X1F = allocF((size_t)B * IC);
    float*  CAH = allocF((size_t)B * 24);
    float*  ATF = allocF((size_t)B * D);   // -> ct (element-wise aliased)
    float*  AIF = allocF((size_t)B * D);   // -> ci

    auto cvt = [&](const float* s, __bf16* d, int n) {
        k_cvt<<<256, 256, 0, stream>>>(s, d, n);
    };
    auto gemm = [&](const __bf16* A, const __bf16* W, int M, int N, int K,
                    const float* bias, const float* g, const float* b,
                    const float* aux, const float* sc, int act,
                    __bf16* ob, float* of) {
        dim3 grid(M / BM, N / BN);
        gemm_bf16_wmma<<<grid, 256, 0, stream>>>(A, W, M, N, K, bias, g, b,
                                                 aux, sc, act, ob, of);
    };

    // --- weight conversion ---
    cvt(tl_w, W_TL, D * D);
    cvt(il_w, W_IL, D * D);
    cvt(sda_wv, W_SV, S * D);
    cvt(sda_wo, W_SO, D * S);
    k_gather_cvt<<<256, 256, 0, stream>>>(fda_w1, W_C1, IC * D, 9, 4);  // center tap
    cvt(fda_wf, W_WF, D * IC);
    cvt(dmi_wq, W_Q, D * D);
    cvt(dmi_wk, W_K, D * D);
    cvt(dmi_wv, W_V, D * D);
    cvt(tg_w1, W_TG1, D * D);
    cvt(tg_w2, W_TG2, D * D);
    cvt(ig_w1, W_IG1, D * D);
    cvt(ig_w2, W_IG2, D * D);
    k_gather_f32<<<2, 256, 0, stream>>>(sa_w, SAW, IC, 49, 24);         // 7x7 center
    cvt(text, TB, B * D);
    cvt(image, IB, B * D);

    // --- stems ---
    gemm(TB, W_TL, B, D, D, tl_b, 0, 0, 0, 0, 1, T1, 0);   // t = GELU(...)
    gemm(IB, W_IL, B, D, D, il_b, 0, 0, 0, 0, 1, IM1, 0);  // im = GELU(...)

    // --- self-attention (seq len 1 -> softmax==1 -> att@v == v) ---
    gemm(T1, W_SV, B, S, D, sda_bv, 0, 0, 0, 0, 0, TV, 0);
    gemm(TV, W_SO, B, D, S, sda_bo, 0, 0, 0, 0, 0, T2, 0);

    // --- FDA branch (all convs on 1x1 spatial -> GEMMs/dots) ---
    gemm(IM1, W_C1, B, IC, D, fda_b1, bn1_g, bn1_b, 0, 0, 2, 0, X1F);
    {
        int n = B * 24;
        k_ca<<<(n + 255) / 256, 256, 0, stream>>>(X1F, ca_w1, CAH, B);
    }
    k_fda_fuse<<<B, 128, 0, stream>>>(X1F, CAH, ca_w2, SAW, sa_b, dec_w,
                                      dec_b, sigma, X4);
    gemm(X4, W_WF, B, D, IC, fda_bf, bn2_g, bn2_b, 0, 0, 2, IM2, 0);

    // --- dmi projections (weights shared across both directions) ---
    __bf16* QPT = TB;   // reuse
    __bf16* KPI = IB;
    __bf16* VPI = T1;
    __bf16* QPI = IM1;
    gemm(T2,  W_Q, B, D, D, dmi_bq, 0, 0, 0, 0, 0, QPT, 0);
    gemm(IM2, W_K, B, D, D, dmi_bk, 0, 0, 0, 0, 0, KPI, 0);
    gemm(IM2, W_V, B, D, D, dmi_bv, 0, 0, 0, 0, 0, VPI, 0);
    gemm(IM2, W_Q, B, D, D, dmi_bq, 0, 0, 0, 0, 0, QPI, 0);
    gemm(T2,  W_K, B, D, D, dmi_bk, 0, 0, 0, 0, 0, KPT, 0);
    gemm(T2,  W_V, B, D, D, dmi_bv, 0, 0, 0, 0, 0, VPT, 0);

    // --- 4096x4096 cross-attention (flash-style), scale = 1/sqrt(768) ---
    const float ascale = 0.03608439182435161f;
    attn_wmma<<<B / 16, 256, 0, stream>>>(QPT, KPI, VPI, ATF, QPT, B, ascale);
    attn_wmma<<<B / 16, 256, 0, stream>>>(QPI, KPT, VPT, AIF, QPI, B, ascale);

    // --- gates:  sigmoid(sigmoid(relu(a@w1+b1)@w2+b2)) * a * scale ---
    __bf16* G1T = KPI;  // reuse
    __bf16* G1I = KPT;
    gemm(QPT, W_TG1, B, D, D, tg_b1, 0, 0, 0, 0, 3, G1T, 0);
    gemm(G1T, W_TG2, B, D, D, tg_b2, 0, 0, ATF, tscale, 4, 0, ATF);  // ct
    gemm(QPI, W_IG1, B, D, D, ig_b1, 0, 0, 0, 0, 3, G1I, 0);
    gemm(G1I, W_IG2, B, D, D, ig_b2, 0, 0, AIF, iscale, 4, 0, AIF);  // ci

    // --- classifier head + softmax ---
    k_head<<<32, 256, 0, stream>>>(ATF, AIF, cls_w, cls_b, (float*)d_out, B);
}
